// LRU_real_12910671691829
// MI455X (gfx1250) — matrix operand order repaired and checked by reading, not compiled, not run on hardware
//
#include <hip/hip_runtime.h>

#define L_SEQ   32768
#define DM      1024
#define DH      1024
#define CHUNK   256
#define NCHUNK  (L_SEQ / CHUNK)

typedef __attribute__((ext_vector_type(16))) __bf16         v16bf;
typedef __attribute__((ext_vector_type(8)))  float          v8f;
typedef __attribute__((ext_vector_type(4)))  unsigned int   u32x4;
typedef __attribute__((ext_vector_type(4)))  float          f32x4;
typedef __attribute__((ext_vector_type(4)))  unsigned short u16x4;

union Frag { v16bf v; u32x4 q[2]; };

__device__ __forceinline__ unsigned short f32_to_bf16(float f) {
  unsigned int u = __float_as_uint(f);
  u += 0x7FFFu + ((u >> 16) & 1u);          // round-to-nearest-even
  return (unsigned short)(u >> 16);
}
__device__ __forceinline__ float bf16_to_f32(unsigned short s) {
  return __uint_as_float(((unsigned int)s) << 16);
}

// ---------------------------------------------------------------- converts
__global__ __launch_bounds__(256) void cvt_f32_bf16(const f32x4* __restrict__ in,
                                                    u16x4* __restrict__ out, int n4) {
  int i = blockIdx.x * 256 + threadIdx.x;
  if (i >= n4) return;
  f32x4 x = in[i];
  u16x4 y;
  y.x = f32_to_bf16(x.x); y.y = f32_to_bf16(x.y);
  y.z = f32_to_bf16(x.z); y.w = f32_to_bf16(x.w);
  out[i] = y;
}

// ---------------------------------------------------------------- WMMA GEMM (NT)
// A: [M,K] bf16 row-major, Bm: [N,K] bf16 row-major (K contiguous for both).
// Per wave: 32 (M) x 128 (N) output strip -> each B fragment feeds 2 WMMAs.
// MODE 0: Out = A*Bm^T as bf16.   MODE 1: Out = A*Bm^T + Dvec[col]*Xres as f32.
template <int MODE>
__global__ __launch_bounds__(256) void wmma_gemm_nt(
    const unsigned short* __restrict__ A,
    const unsigned short* __restrict__ Bm,
    unsigned short* __restrict__ OutBf,
    float* __restrict__ OutF,
    const float* __restrict__ Dvec,
    const float* __restrict__ Xres,
    int N, int K)
{
  const int lane   = threadIdx.x & 31;
  const int wave   = threadIdx.x >> 5;
  const int m_base = (blockIdx.x * 8 + wave) * 32;   // 32 rows per wave
  const int n_base = blockIdx.y * 128;
  const int lo16   = lane & 15;
  const int khalf  = lane >> 4;       // 0 or 1

  v8f acc[2][8];
#pragma unroll
  for (int mt = 0; mt < 2; ++mt)
#pragma unroll
    for (int i = 0; i < 8; ++i)
#pragma unroll
      for (int j = 0; j < 8; ++j) acc[mt][i][j] = 0.0f;

  const unsigned short* Arow0 = A + (size_t)(m_base + lo16) * K;
  const unsigned short* Arow1 = Arow0 + (size_t)16 * K;

  for (int k = 0; k < K; k += 32) {
    Frag a0, a1;
    // A-frag: lanes 0-15 hold K {0..7,16..23}, lanes 16-31 hold K {8..15,24..31}
    a0.q[0] = *(const u32x4*)(Arow0 + k + khalf * 8);
    a0.q[1] = *(const u32x4*)(Arow0 + k + 16 + khalf * 8);
    a1.q[0] = *(const u32x4*)(Arow1 + k + khalf * 8);
    a1.q[1] = *(const u32x4*)(Arow1 + k + 16 + khalf * 8);
#pragma unroll
    for (int nt = 0; nt < 8; ++nt) {
      // B-frag: lane holds column n = lo16, K {khalf*8..+7, 16+khalf*8..+7} (same as A)
      const unsigned short* bp =
          Bm + (size_t)(n_base + nt * 16 + lo16) * K + k + khalf * 8;
      Frag b;
      b.q[0] = *(const u32x4*)(bp);
      b.q[1] = *(const u32x4*)(bp + 16);
      acc[0][nt] = __builtin_amdgcn_wmma_f32_16x16x32_bf16(
          false, a0.v, false, b.v, (short)0, acc[0][nt], false, false);
      acc[1][nt] = __builtin_amdgcn_wmma_f32_16x16x32_bf16(
          false, a1.v, false, b.v, (short)0, acc[1][nt], false, false);
    }
  }

  // C/D layout: VGPR r -> row = r + 8*(lane>=16), col = lane&15
  const int row_off = khalf * 8;
#pragma unroll
  for (int mt = 0; mt < 2; ++mt) {
#pragma unroll
    for (int nt = 0; nt < 8; ++nt) {
#pragma unroll
      for (int r = 0; r < 8; ++r) {
        int row = m_base + mt * 16 + r + row_off;
        int col = n_base + nt * 16 + lo16;
        size_t idx = (size_t)row * N + col;
        if (MODE == 0) {
          OutBf[idx] = f32_to_bf16(acc[mt][nt][r]);
        } else {
          OutF[idx] = acc[mt][nt][r] + Dvec[col] * Xres[idx];
        }
      }
    }
  }
}

// ---------------------------------------------------------------- chunked scan
// Pass 1: per-chunk local scan (zero init), in-place bf16, fp32 carry out.
__global__ __launch_bounds__(256) void scan_local(unsigned short* __restrict__ bu,
                                                  const float* __restrict__ lambda,
                                                  float* __restrict__ carry) {
  int h = blockIdx.y * 256 + threadIdx.x;
  int c = blockIdx.x;
  float lam = lambda[h];
  size_t idx = (size_t)c * CHUNK * DH + h;
  float s = 0.0f;
  for (int t = 0; t < CHUNK; ++t) {
    s = lam * s + bf16_to_f32(bu[idx]);
    bu[idx] = f32_to_bf16(s);
    idx += DH;
  }
  carry[(size_t)c * DH + h] = s;
}

// Pass 2: per-channel exclusive prefix of chunk carries: p_{c+1} = lam^CHUNK * p_c + carry_c
__global__ __launch_bounds__(256) void scan_prefix(const float* __restrict__ carry,
                                                   float* __restrict__ prefix,
                                                   const float* __restrict__ lambda) {
  int h = blockIdx.x * 256 + threadIdx.x;
  float lamT = lambda[h];
#pragma unroll
  for (int i = 0; i < 8; ++i) lamT *= lamT;   // lambda^256 (CHUNK = 2^8)
  float p = 0.0f;
  for (int c = 0; c < NCHUNK; ++c) {
    prefix[(size_t)c * DH + h] = p;
    p = lamT * p + carry[(size_t)c * DH + h];
  }
}

// Pass 3: h_global(t) = local(t) + lambda^{t+1} * p_c  (chunk 0 needs no fixup)
__global__ __launch_bounds__(256) void scan_apply(unsigned short* __restrict__ hid,
                                                  const float* __restrict__ lambda,
                                                  const float* __restrict__ prefix) {
  int h = blockIdx.y * 256 + threadIdx.x;
  int c = blockIdx.x + 1;
  float lam = lambda[h];
  float p   = prefix[(size_t)c * DH + h];
  float pw  = lam;
  size_t idx = (size_t)c * CHUNK * DH + h;
  for (int t = 0; t < CHUNK; ++t) {
    float v = bf16_to_f32(hid[idx]) + pw * p;
    hid[idx] = f32_to_bf16(v);
    pw *= lam;
    idx += DH;
  }
}

// ---------------------------------------------------------------- launch
extern "C" void kernel_launch(void* const* d_in, const int* in_sizes, int n_in,
                              void* d_out, int out_size, void* d_ws, size_t ws_size,
                              hipStream_t stream) {
  (void)in_sizes; (void)n_in; (void)out_size; (void)ws_size;
  const float* x   = (const float*)d_in[0];   // [L, DM]
  const float* lam = (const float*)d_in[1];   // [DH]
  const float* B   = (const float*)d_in[2];   // [DH, DM]
  const float* C   = (const float*)d_in[3];   // [DM, DH]
  const float* Dv  = (const float*)d_in[4];   // [DM]
  float* out = (float*)d_out;                 // [L, DM] f32

  char* ws = (char*)d_ws;
  unsigned short* Xbf = (unsigned short*)(ws);                 // 64 MB bf16 X
  unsigned short* Hbf = (unsigned short*)(ws + 67108864);      // 64 MB Bu -> hidden (in place)
  unsigned short* Bbf = (unsigned short*)(ws + 134217728);     //  2 MB
  unsigned short* Cbf = (unsigned short*)(ws + 136314880);     //  2 MB
  float* carry  = (float*)(ws + 138412032);                    // 512 KB
  float* prefix = (float*)(ws + 138936320);                    // 512 KB

  // fp32 -> bf16 operand conversion
  cvt_f32_bf16<<<(L_SEQ * DM / 4 + 255) / 256, 256, 0, stream>>>(
      (const f32x4*)x, (u16x4*)Xbf, L_SEQ * DM / 4);
  cvt_f32_bf16<<<(DH * DM / 4 + 255) / 256, 256, 0, stream>>>(
      (const f32x4*)B, (u16x4*)Bbf, DH * DM / 4);
  cvt_f32_bf16<<<(DM * DH / 4 + 255) / 256, 256, 0, stream>>>(
      (const f32x4*)C, (u16x4*)Cbf, DM * DH / 4);

  // GEMM 1: Bu = X * B^T   [L, DH] bf16   (block: 256 M x 128 N)
  dim3 g1(L_SEQ / 256, DH / 128);
  wmma_gemm_nt<0><<<g1, 256, 0, stream>>>(Xbf, Bbf, Hbf, nullptr, nullptr, nullptr, DH, DM);

  // chunked associative scan: h_t = lam*h_{t-1} + Bu_t (in place in Hbf)
  scan_local <<<dim3(NCHUNK, DH / 256), 256, 0, stream>>>(Hbf, lam, carry);
  scan_prefix<<<DH / 256, 256, 0, stream>>>(carry, prefix, lam);
  scan_apply <<<dim3(NCHUNK - 1, DH / 256), 256, 0, stream>>>(Hbf, lam, prefix);

  // GEMM 2: out = H * C^T + D .* X   [L, DM] f32
  dim3 g2(L_SEQ / 256, DM / 128);
  wmma_gemm_nt<1><<<g2, 256, 0, stream>>>(Hbf, Cbf, nullptr, out, Dv, x, DM, DH);
}
